// HAMP_43585328120461
// MI455X (gfx1250) — matrix operand order repaired and checked by reading, not compiled, not run on hardware
//
#include <hip/hip_runtime.h>

// ---------------------------------------------------------------------------
// Problem dimensions (from the reference)
// ---------------------------------------------------------------------------
constexpr int TT   = 2;
constexpr int NN   = 16384;
constexpr int EE   = 262144;
constexpr int INF  = 128;
constexpr int HID  = 256;
constexpr int NH   = 8;     // heads
constexpr int DK   = 32;
constexpr int NL   = 2;     // layers
constexpr int OUTD = 64;

typedef __attribute__((ext_vector_type(16))) __bf16    v16bf;
typedef __attribute__((ext_vector_type(8)))  float     v8f;
typedef __attribute__((ext_vector_type(4)))  unsigned  v4u;
typedef __attribute__((ext_vector_type(8)))  int       v8i;
typedef __attribute__((ext_vector_type(4)))  int       v4i;

__device__ __forceinline__ float gelu_f(float x) {
    return 0.5f * x * (1.0f + erff(x * 0.70710678118654752f));
}

// order-preserving float<->uint encoding for atomicMax-based segment max
__device__ __forceinline__ unsigned enc_f32(float f) {
    unsigned u = __float_as_uint(f);
    return (u & 0x80000000u) ? ~u : (u | 0x80000000u);
}
__device__ __forceinline__ float dec_f32(unsigned k) {
    unsigned u = (k & 0x80000000u) ? (k & 0x7FFFFFFFu) : ~k;
    return __uint_as_float(u);
}

// ---------------------------------------------------------------------------
// Tensor Data Mover: 2D tile load Global->LDS (D# per ISA 08_async_tensor §8).
//   group0: [0]=count=1, [1]=lds byte addr, [2]=global_addr[31:0],
//           [3]=global_addr[56:32] | type=2<<30
//   group1: [0]=data_size(4B)=2<<16, dims/tiles/stride packed per spec.
//   groups 2/3/4 zero (2D tensor; VADDR2/3 disabled semantics).
// tensor_dim0 < tile_dim0 => TDM zero-fills out-of-range columns (OOB-safe).
// This toolchain uses the 6-arg builtin form:
//   (v4u g0, v8i g1, v4i g2, v4i g3, v8i g4, i32 cpol)
// ---------------------------------------------------------------------------
__device__ __forceinline__ void tdm_load_2d(unsigned lds_off, const void* gptr,
                                            int tile0, int tile1,
                                            int tdim0, int tdim1,
                                            long long stride0)
{
    unsigned long long ga = (unsigned long long)gptr;
    v4u g0 = { 1u,
               lds_off,
               (unsigned)(ga & 0xFFFFFFFFull),
               (unsigned)((ga >> 32) & 0x1FFFFFFull) | (2u << 30) };
    v8i g1;
    g1[0] = (int)(2u << 16);                                            // data_size = 4B
    g1[1] = (int)(((unsigned)tdim0 & 0xFFFFu) << 16);                   // tensor_dim0 lo
    g1[2] = (int)((((unsigned)tdim0 >> 16) & 0xFFFFu)
                  | (((unsigned)tdim1 & 0xFFFFu) << 16));               // dim0 hi | dim1 lo
    g1[3] = (int)((((unsigned)tdim1 >> 16) & 0xFFFFu)
                  | (((unsigned)tile0 & 0xFFFFu) << 16));               // dim1 hi | tile0
    g1[4] = (int)((unsigned)tile1 & 0xFFFFu);                           // tile1 (tile2=0)
    g1[5] = (int)(unsigned)(stride0 & 0xFFFFFFFFll);                    // dim0 stride lo
    g1[6] = (int)((stride0 >> 32) & 0xFFFFll);                          // dim0 stride hi
    g1[7] = 0;
    v4i g2 = {0, 0, 0, 0};
    v4i g3 = {0, 0, 0, 0};
    v8i g4 = {0, 0, 0, 0, 0, 0, 0, 0};
    __builtin_amdgcn_tensor_load_to_lds(g0, g1, g2, g3, g4, 0);
}

// ---------------------------------------------------------------------------
// Generic GEMM: C[M,Nc] = act(A[M,K] @ B[K,Nc] + bias), fp32 in/out,
// bf16 WMMA f32-accumulate core, TDM-staged LDS operand tiles.
//
// Block = 256 threads = 8 waves, block tile 64x64.  Wave w owns the 16x32
// strip at (m_w = (w>>1)*16, n_w = (w&1)*32): one A fragment, two B
// fragments, two v_wmma_f32_16x16x32_bf16 per K-step.
// Per K-step, wave 0 issues two tensor_load_to_lds (A: 64x32, B: 32x64),
// waits TENSORcnt, then a workgroup barrier publishes the tiles.
// Requires M % 64 == 0 and K % 32 == 0 (true at every call site);
// Nc handled via TDM zero-fill + store guard.
//
// Fragment layouts per CDNA5 ISA 7.12.2:
//   A (16x32 bf16): row = lane%16; elem e -> k = a_koff + (e<8 ? e : e+8),
//                   a_koff = (lane>=16) ? 8 : 0.
//   B (32x16 bf16): col = lane%16; elem e -> k = b_koff + (e<8 ? e : e+8),
//                   b_koff = (lane>=16) ? 8 : 0.
//   C (16x16 f32):  vgpr r -> row = r + (lane>=16 ? 8 : 0), col = lane%16.
// ---------------------------------------------------------------------------
__global__ __launch_bounds__(256) void gemm_bf16_tdm_k(
    const float* __restrict__ A, const float* __restrict__ B,
    const float* __restrict__ bias, float* __restrict__ C,
    int M, int Nc, int K, int lda, int ldb, int ldc, int act,
    const int* __restrict__ selA, long long selStride)
{
    __shared__ float Alds[64 * 32];
    __shared__ float Blds[32 * 64];

    if (selA) A += (long long)selA[0] * selStride;   // device-side row-block select

    const int lane = threadIdx.x & 31;
    const int wave = threadIdx.x >> 5;
    const int blkm = blockIdx.x * 64;
    const int blkn = blockIdx.y * 64;
    const int m_w  = (wave >> 1) * 16;
    const int n_w  = (wave & 1) * 32;

    const int fr     = lane & 15;
    const int a_koff = (lane >= 16) ? 8 : 0;
    const int b_koff = (lane >= 16) ? 8 : 0;

    const unsigned lds_a = (unsigned)(unsigned long long)(void*)Alds;
    const unsigned lds_b = (unsigned)(unsigned long long)(void*)Blds;
    const int bvalid = Nc - blkn;     // valid B columns at this block (>=1)

    v8f acc0 = {0.f, 0.f, 0.f, 0.f, 0.f, 0.f, 0.f, 0.f};
    v8f acc1 = {0.f, 0.f, 0.f, 0.f, 0.f, 0.f, 0.f, 0.f};

    const float* a_lrow = Alds + (m_w + fr) * 32 + a_koff;

    for (int k0 = 0; k0 < K; k0 += 32) {
        if (wave == 0) {
            tdm_load_2d(lds_a, A + (long long)blkm * lda + k0,
                        /*tile0=*/32, /*tile1=*/64, /*tdim0=*/32, /*tdim1=*/64,
                        (long long)lda);
            tdm_load_2d(lds_b, B + (long long)k0 * ldb + blkn,
                        /*tile0=*/64, /*tile1=*/32, /*tdim0=*/bvalid, /*tdim1=*/32,
                        (long long)ldb);
            __builtin_amdgcn_s_wait_tensorcnt(0);
        }
        __syncthreads();                // tiles visible to all waves

        v16bf a, b0, b1;
#pragma unroll
        for (int e = 0; e < 8; ++e)  a[e] = (__bf16)a_lrow[e];
#pragma unroll
        for (int e = 8; e < 16; ++e) a[e] = (__bf16)a_lrow[e + 8];
#pragma unroll
        for (int e = 0; e < 16; ++e) {
            const float* brow = Blds + (b_koff + e + (e >= 8 ? 8 : 0)) * 64;
            b0[e] = (__bf16)brow[n_w + fr];
            b1[e] = (__bf16)brow[n_w + 16 + fr];
        }
        acc0 = __builtin_amdgcn_wmma_f32_16x16x32_bf16(
            false, a, false, b0, (short)0, acc0, false, false);
        acc1 = __builtin_amdgcn_wmma_f32_16x16x32_bf16(
            false, a, false, b1, (short)0, acc1, false, false);

        __syncthreads();                // all reads done before next overwrite
    }

    const int c_roff = (lane >= 16) ? 8 : 0;
    const int row0   = blkm + m_w + c_roff;
    const int col0   = blkn + n_w + fr;
    const int col1   = col0 + 16;
    const float bv0  = (bias && col0 < Nc) ? bias[col0] : 0.f;
    const float bv1  = (bias && col1 < Nc) ? bias[col1] : 0.f;
#pragma unroll
    for (int r = 0; r < 8; ++r) {
        if (col0 < Nc) {
            float x = acc0[r] + bv0;
            if (act == 1) x = gelu_f(x);
            C[(long long)(row0 + r) * ldc + col0] = x;
        }
        if (col1 < Nc) {
            float x = acc1[r] + bv1;
            if (act == 1) x = gelu_f(x);
            C[(long long)(row0 + r) * ldc + col1] = x;
        }
    }
}

// ---------------------------------------------------------------------------
// posvect activation materialization: abuf[n, kk] for one (t,m).
// kk = g*256 + hh, g selects {sin, linear, softplus, sigmoid} branch.
// ---------------------------------------------------------------------------
__global__ __launch_bounds__(256) void pv_act_k(
    const float* __restrict__ scalars,
    const float* __restrict__ ww, const float* __restrict__ wb,
    const float* __restrict__ lw, const float* __restrict__ lb,
    const float* __restrict__ sw, const float* __restrict__ sb,
    const float* __restrict__ gw, const float* __restrict__ gb,
    float* __restrict__ abuf, int t, int m)
{
    int i = blockIdx.x * 256 + threadIdx.x;       // over NN*1024
    int n  = i >> 10;
    int kk = i & 1023;
    int g  = kk >> 8;
    int hh = kk & 255;
    float x = scalars[((long long)t * NN + n) * 6 + m];
    long long widx = ((long long)t * 6 + m) * 256 + hh;
    float w, b;
    if      (g == 0) { w = ww[widx]; b = wb[widx]; }
    else if (g == 1) { w = lw[widx]; b = lb[widx]; }
    else if (g == 2) { w = sw[widx]; b = sb[widx]; }
    else             { w = gw[widx]; b = gb[widx]; }
    float v = fmaf(x, w, b);
    float r;
    if      (g == 0) r = __sinf(v);
    else if (g == 1) r = v;
    else if (g == 2) r = (v > 20.f) ? v : log1pf(__expf(v));
    else             r = 1.f / (1.f + __expf(-v));
    abuf[i] = r;
}

// ---------------------------------------------------------------------------
// ima mixing: per node, w_m = tanh(z_m @ w1 + b1) @ w2, softmax over the 7
// components, h = sum_m p_m * z_m. One 256-thread block per (t,n); z staged
// in LDS; waves 0..6 own component m with __shfl_xor reduction.
// ---------------------------------------------------------------------------
__global__ __launch_bounds__(256) void ima_k(
    const float* __restrict__ z, const float* __restrict__ w1,
    const float* __restrict__ b1, const float* __restrict__ w2,
    float* __restrict__ h)
{
    __shared__ float zs[7 * 256];
    __shared__ float sc[7];
    long long tn = blockIdx.x;                   // t*NN + n
    const float* zrow = z + tn * (7 * 256);
    for (int i = threadIdx.x; i < 7 * 256; i += 256) zs[i] = zrow[i];
    __syncthreads();

    int wave = threadIdx.x >> 5, lane = threadIdx.x & 31;
    if (wave < 7) {
        const int m = wave;
        float part = 0.f;
#pragma unroll
        for (int kb = 0; kb < 4; ++kb) {
            int k = lane + kb * 32;
            float acc = b1[k];
            for (int hh = 0; hh < 256; ++hh)
                acc = fmaf(zs[m * 256 + hh], w1[hh * 128 + k], acc);
            part = fmaf(tanhf(acc), w2[k], part);
        }
        for (int o = 16; o > 0; o >>= 1) part += __shfl_xor(part, o);
        if (lane == 0) sc[m] = part;
    }
    __syncthreads();

    float mx = sc[0];
#pragma unroll
    for (int m = 1; m < 7; ++m) mx = fmaxf(mx, sc[m]);
    float p[7], sum = 0.f;
#pragma unroll
    for (int m = 0; m < 7; ++m) { p[m] = __expf(sc[m] - mx); sum += p[m]; }
    float inv = 1.f / sum;
    int c = threadIdx.x;
    float acc = 0.f;
#pragma unroll
    for (int m = 0; m < 7; ++m) acc = fmaf(p[m], zs[m * 256 + c], acc);
    h[tn * 256 + c] = acc * inv;
}

// ---------------------------------------------------------------------------
// Edge attention: thread per (edge, head).
// ---------------------------------------------------------------------------
__global__ __launch_bounds__(256) void edge_score_k(
    const int* __restrict__ src, const int* __restrict__ dst,
    const float* __restrict__ q, const float* __restrict__ ke,
    const float* __restrict__ canon, float* __restrict__ score,
    unsigned* __restrict__ segmax, int l, int e)
{
    int idx = blockIdx.x * 256 + threadIdx.x;
    if (idx >= EE * NH) return;
    int edge = idx >> 3, hh = idx & 7;
    int s = src[e * EE + edge];
    int d = dst[e * EE + edge];
    int dt = (e == 0) ? 1 : 0;
    const float* qrow = q  + ((long long)dt * NN + d) * 256 + hh * 32;
    const float* krow = ke + (long long)s * 256 + hh * 32;
    float dotv = 0.f;
#pragma unroll
    for (int i = 0; i < 32; ++i) dotv = fmaf(qrow[i], krow[i], dotv);
    float scv = dotv * canon[(l * 2 + e) * NH + hh] * 0.17677669529663689f; // 1/sqrt(32)
    score[idx] = scv;
    atomicMax(segmax + d * NH + hh, enc_f32(scv));
}

__global__ __launch_bounds__(256) void edge_exp_k(
    const int* __restrict__ dst, float* __restrict__ score,
    const unsigned* __restrict__ segmax, float* __restrict__ segsum, int e)
{
    int idx = blockIdx.x * 256 + threadIdx.x;
    if (idx >= EE * NH) return;
    int edge = idx >> 3, hh = idx & 7;
    int d = dst[e * EE + edge];
    float m  = dec_f32(segmax[d * NH + hh]);
    float ev = __expf(score[idx] - m);
    score[idx] = ev;
    atomicAdd(segsum + d * NH + hh, ev);
}

__global__ __launch_bounds__(256) void edge_agg_k(
    const int* __restrict__ src, const int* __restrict__ dst,
    const float* __restrict__ score, const float* __restrict__ segsum,
    const float* __restrict__ ve, float* __restrict__ tb, int e)
{
    int idx = blockIdx.x * 256 + threadIdx.x;
    if (idx >= EE * NH) return;
    int edge = idx >> 3, hh = idx & 7;
    int s = src[e * EE + edge];
    int d = dst[e * EE + edge];
    int dt = (e == 0) ? 1 : 0;
    float p = score[idx] / segsum[d * NH + hh];
    const float* vrow = ve + (long long)s * 256 + hh * 32;
    float* arow = tb + ((long long)dt * NN + d) * 256 + hh * 32;
#pragma unroll
    for (int i = 0; i < 32; ++i) atomicAdd(arow + i, p * vrow[i]);
}

// ---------------------------------------------------------------------------
// Fused residual gate + LayerNorm: h = LN(hp*alpha + h*(1-alpha)).
// One 256-thread block per row.
// ---------------------------------------------------------------------------
__global__ __launch_bounds__(256) void fuse_ln_k(
    const float* __restrict__ hp, float* __restrict__ h,
    const float* __restrict__ res, const float* __restrict__ ln_g,
    const float* __restrict__ ln_b, int l)
{
    int row = blockIdx.x;                // t*NN + n
    int t = row >> 14;                   // NN = 2^14
    int c = threadIdx.x;
    float alpha = 1.f / (1.f + __expf(-res[l * TT + t]));
    long long base = (long long)row * 256;
    float v = hp[base + c] * alpha + h[base + c] * (1.f - alpha);

    float s = v, s2 = v * v;
    for (int o = 16; o > 0; o >>= 1) { s += __shfl_xor(s, o); s2 += __shfl_xor(s2, o); }
    __shared__ float sh1[8], sh2[8];
    int wave = c >> 5, lane = c & 31;
    if (lane == 0) { sh1[wave] = s; sh2[wave] = s2; }
    __syncthreads();
    float t1 = 0.f, t2 = 0.f;
#pragma unroll
    for (int w = 0; w < 8; ++w) { t1 += sh1[w]; t2 += sh2[w]; }
    float mu  = t1 * (1.f / 256.f);
    float var = t2 * (1.f / 256.f) - mu * mu;
    long long gi = (long long)(l * TT + t) * 256 + c;
    h[base + c] = (v - mu) * rsqrtf(var + 1e-5f) * ln_g[gi] + ln_b[gi];
}

// ---------------------------------------------------------------------------
// Host-side launch helpers
// ---------------------------------------------------------------------------
static inline void gemm(hipStream_t s, const float* A, const float* B,
                        const float* bias, float* C, int M, int Nc, int K,
                        int lda, int ldb, int ldc, int act,
                        const int* sel = nullptr, long long selStride = 0)
{
    dim3 g((M + 63) / 64, (Nc + 63) / 64);
    gemm_bf16_tdm_k<<<g, dim3(256), 0, s>>>(A, B, bias, C, M, Nc, K,
                                            lda, ldb, ldc, act, sel, selStride);
}

extern "C" void kernel_launch(void* const* d_in, const int* in_sizes, int n_in,
                              void* d_out, int out_size, void* d_ws, size_t ws_size,
                              hipStream_t stream)
{
    (void)in_sizes; (void)n_in; (void)out_size; (void)ws_size;

    const float* node_ft  = (const float*)d_in[0];
    const float* scalars  = (const float*)d_in[1];
    const int*   src_idx  = (const int*)  d_in[2];
    const int*   dst_idx  = (const int*)  d_in[3];
    const float* proj_w   = (const float*)d_in[4];
    const float* proj_b   = (const float*)d_in[5];
    const float* pv_ww    = (const float*)d_in[6];
    const float* pv_wb    = (const float*)d_in[7];
    const float* pv_lw    = (const float*)d_in[8];
    const float* pv_lb    = (const float*)d_in[9];
    const float* pv_sw    = (const float*)d_in[10];
    const float* pv_sb    = (const float*)d_in[11];
    const float* pv_gw    = (const float*)d_in[12];
    const float* pv_gb    = (const float*)d_in[13];
    const float* pv_fc_w  = (const float*)d_in[14];
    const float* pv_fc_b  = (const float*)d_in[15];
    const float* ima_w1   = (const float*)d_in[16];
    const float* ima_b1   = (const float*)d_in[17];
    const float* ima_w2   = (const float*)d_in[18];
    const float* lyr_k_w  = (const float*)d_in[19];
    const float* lyr_q_w  = (const float*)d_in[20];
    const float* lyr_v_w  = (const float*)d_in[21];
    const float* lyr_fc_w = (const float*)d_in[22];
    const float* lyr_k_b  = (const float*)d_in[23];
    const float* lyr_q_b  = (const float*)d_in[24];
    const float* lyr_v_b  = (const float*)d_in[25];
    const float* lyr_fc_b = (const float*)d_in[26];
    const float* lyr_att_w= (const float*)d_in[27];
    const float* lyr_val_w= (const float*)d_in[28];
    const float* lyr_canon= (const float*)d_in[29];
    const float* lyr_res  = (const float*)d_in[30];
    const float* lyr_ln_g = (const float*)d_in[31];
    const float* lyr_ln_b = (const float*)d_in[32];
    const float* out_w    = (const float*)d_in[33];
    const float* out_b    = (const float*)d_in[34];
    const int*   sel      = (const int*)  d_in[35];

    // Workspace layout (floats).  z = [T,N,7,HID] is dead after ima_k, so all
    // layer-loop scratch is aliased inside it.  Total ws need ~336 MB.
    float* ws   = (float*)d_ws;
    float* z    = ws;                                  // 58,720,256 f
    float* h    = ws + 58720256;                       //  8,388,608 f
    float* abuf = ws + 58720256 + 8388608;             // 16,777,216 f
    // aliases into z (used only after ima):
    float*    kb     = z;
    float*    qb     = z + 8388608;
    float*    vb     = z + 16777216;
    float*    tb     = z + 25165824;                   // [2,N,HID] agg
    float*    hp     = z + 33554432;
    float*    keb    = z + 41943040;                   // [N,HID]
    float*    veb    = z + 46137344;                   // [N,HID]
    float*    score  = z + 50331648;                   // [E*H]
    unsigned* segmax = (unsigned*)(z + 52428800);      // [N*H]
    float*    segsum = z + 52559872;                   // [N*H]

    // ---- Stage 1: h0 = gelu(node_ft @ proj_w + b) -> z[:,:,0,:] ------------
    for (int t = 0; t < TT; ++t) {
        gemm(stream,
             node_ft + (long long)t * NN * INF, proj_w + (long long)t * INF * HID,
             proj_b + t * HID,
             z + ((long long)t * NN * 7 + 0) * HID,
             NN, HID, INF, INF, HID, 7 * HID, /*gelu*/1);
    }

    // ---- Stage 2: posvect: 12 x [N,1024]@[1024,256] -> z[:,:,1+m,:] -------
    for (int t = 0; t < TT; ++t) {
        for (int m = 0; m < 6; ++m) {
            pv_act_k<<<dim3((NN * 1024) / 256), dim3(256), 0, stream>>>(
                scalars, pv_ww, pv_wb, pv_lw, pv_lb, pv_sw, pv_sb, pv_gw, pv_gb,
                abuf, t, m);
            gemm(stream, abuf,
                 pv_fc_w + ((long long)t * 6 + m) * 1024 * HID,
                 pv_fc_b + ((long long)t * 6 + m) * HID,
                 z + ((long long)t * NN * 7 + (1 + m)) * HID,
                 NN, HID, 1024, 1024, HID, 7 * HID, /*gelu*/1);
        }
    }

    // ---- Stage 3: ima attention over 7 components -> h ---------------------
    ima_k<<<dim3(TT * NN), dim3(256), 0, stream>>>(z, ima_w1, ima_b1, ima_w2, h);

    // ---- Stage 4: transformer layers ---------------------------------------
    for (int l = 0; l < NL; ++l) {
        // k,q,v projections
        for (int t = 0; t < TT; ++t) {
            long long wo = (long long)(l * TT + t) * HID * HID;
            long long bo = (long long)(l * TT + t) * HID;
            long long ho = (long long)t * NN * HID;
            gemm(stream, h + ho, lyr_k_w + wo, lyr_k_b + bo, kb + ho,
                 NN, HID, HID, HID, HID, HID, 0);
            gemm(stream, h + ho, lyr_q_w + wo, lyr_q_b + bo, qb + ho,
                 NN, HID, HID, HID, HID, HID, 0);
            gemm(stream, h + ho, lyr_v_w + wo, lyr_v_b + bo, vb + ho,
                 NN, HID, HID, HID, HID, HID, 0);
        }

        (void)hipMemsetAsync(tb, 0, (size_t)2 * NN * HID * sizeof(float), stream);

        for (int e = 0; e < 2; ++e) {
            const int st = (e == 0) ? 0 : 1;
            // per-head ke/ve transforms via WMMA (K=32, Nc=32, TDM zero-fill
            // handles the 64-wide block tile)
            for (int hh = 0; hh < NH; ++hh) {
                long long awo = (long long)((l * 2 + e) * NH + hh) * DK * DK;
                gemm(stream, kb + (long long)st * NN * HID + hh * DK,
                     lyr_att_w + awo, nullptr, keb + hh * DK,
                     NN, DK, DK, HID, DK, HID, 0);
                gemm(stream, vb + (long long)st * NN * HID + hh * DK,
                     lyr_val_w + awo, nullptr, veb + hh * DK,
                     NN, DK, DK, HID, DK, HID, 0);
            }
            (void)hipMemsetAsync(segmax, 0, (size_t)NN * NH * sizeof(unsigned), stream);
            (void)hipMemsetAsync(segsum, 0, (size_t)NN * NH * sizeof(float), stream);

            dim3 eg((EE * NH + 255) / 256);
            edge_score_k<<<eg, dim3(256), 0, stream>>>(
                src_idx, dst_idx, qb, keb, lyr_canon, score, segmax, l, e);
            edge_exp_k<<<eg, dim3(256), 0, stream>>>(dst_idx, score, segmax, segsum, e);
            edge_agg_k<<<eg, dim3(256), 0, stream>>>(src_idx, dst_idx, score, segsum,
                                                     veb, tb, e);
        }

        // fc + gated residual + LayerNorm
        for (int t = 0; t < TT; ++t) {
            long long wo = (long long)(l * TT + t) * HID * HID;
            long long bo = (long long)(l * TT + t) * HID;
            long long ho = (long long)t * NN * HID;
            gemm(stream, tb + ho, lyr_fc_w + wo, lyr_fc_b + bo, hp + ho,
                 NN, HID, HID, HID, HID, HID, 0);
        }
        fuse_ln_k<<<dim3(TT * NN), dim3(256), 0, stream>>>(
            hp, h, lyr_res, lyr_ln_g, lyr_ln_b, l);
    }

    // ---- Stage 5: output head: h[sel] @ out_w + out_b ----------------------
    gemm(stream, h, out_w, out_b, (float*)d_out,
         NN, OUTD, HID, HID, OUTD, OUTD, 0, sel, (long long)NN * HID);
}
